// Decoder_48833778155677
// MI455X (gfx1250) — matrix operand order repaired and checked by reading, not compiled, not run on hardware
//
#include <hip/hip_runtime.h>
#include <hip/hip_bf16.h>

// ---------------------------------------------------------------------------
// Problem constants
// ---------------------------------------------------------------------------
#define VV 32000
#define EE 512
#define HH 1024
#define BB 16
#define TT 128
#define SS 128

typedef __attribute__((ext_vector_type(16))) __bf16 v16bf;
typedef __attribute__((ext_vector_type(8)))  float  v8f;

union FragBF { unsigned int u[8]; v16bf v; };

__device__ __forceinline__ unsigned short f2bf(float x) {
    unsigned int b = __float_as_uint(x);
    unsigned int r = b + 0x7FFFu + ((b >> 16) & 1u);   // round-to-nearest-even
    return (unsigned short)(r >> 16);
}

// A fragment: 16x32 (MxK) bf16, contiguous k = 16*hi + s (same order as B); sA = [16][ldk] ushort.
__device__ __forceinline__ v16bf load_fragA(const unsigned short* sA, int ldk) {
    int lane = threadIdx.x & 31;
    int m = lane & 15, hi = lane >> 4;
    FragBF f;
#pragma unroll
    for (int p = 0; p < 8; ++p) {
        int k = hi * 16 + 2 * p;
        f.u[p] = *(const unsigned int*)(sA + m * ldk + k);
    }
    return f.v;
}

// B fragment: 32x16 (KxN); sB staged as [n][k]; lanes 0-15 K=0..15, 16-31 K=16..31.
__device__ __forceinline__ v16bf load_fragB(const unsigned short* sB, int ldk) {
    int lane = threadIdx.x & 31;
    int n = lane & 15, hi = lane >> 4;
    FragBF f;
#pragma unroll
    for (int p = 0; p < 8; ++p) {
        int k = hi * 16 + 2 * p;
        f.u[p] = *(const unsigned int*)(sB + n * ldk + k);
    }
    return f.v;
}

#define WMMA_BF16(A_, B_, C_) \
    __builtin_amdgcn_wmma_f32_16x16x32_bf16(false, (A_), false, (B_), (short)0, (C_), false, false)

// CDNA5 async global->LDS copy of 16 bytes (tracked by ASYNCcnt).
__device__ __forceinline__ void async_copy_b128(unsigned lds_byte_addr,
                                                unsigned long long global_byte_addr) {
    asm volatile("global_load_async_to_lds_b128 %0, %1, off"
                 :: "v"(lds_byte_addr), "v"(global_byte_addr) : "memory");
}
__device__ __forceinline__ void wait_async() {
    asm volatile("s_wait_asynccnt 0x0" ::: "memory");
}
// Low 32 bits of a generic pointer to LDS == LDS byte address (ISA 10.2 aperture).
__device__ __forceinline__ unsigned lds_addr(const void* p) {
    return (unsigned)(unsigned long long)p;
}

// ---------------------------------------------------------------------------
// Batched GEMM:  D[M,N] = A[M,K](f32) * op(B)(bf16) + bias, optional tanh.
//   BTRANS=1: Bsrc is [N,K] row-major  (weights, async-copied straight to LDS)
//   BTRANS=0: Bsrc is [K,N] row-major  (enc_out for the ctx GEMM)
// 256 threads = 8 waves; block tile 64x64; wave tile 16x32; K-step 64
// -> 4 v_wmma per inner iteration per wave.
// All launch dims are exact multiples of the tiles: no guards anywhere.
// ---------------------------------------------------------------------------
template<int BTRANS, int ACT>
__global__ __launch_bounds__(256)
void gemm_bf16(const float* __restrict__ A, const unsigned short* __restrict__ Bsrc,
               const float* __restrict__ bias, float* __restrict__ D,
               int M, int N, int K, int lda, int ldb, int ldd,
               long strideA, long strideB, long strideD)
{
    __shared__ __attribute__((aligned(16))) unsigned short shA[64 * 64];
    __shared__ __attribute__((aligned(16))) unsigned short shB[64 * 64];

    A    += (long)blockIdx.z * strideA;
    Bsrc += (long)blockIdx.z * strideB;
    D    += (long)blockIdx.z * strideD;

    const int tile_n = blockIdx.x * 64;
    const int tile_m = blockIdx.y * 64;
    const int tid  = threadIdx.x;
    const int lane = tid & 31;
    const int wave = tid >> 5;
    const int wm   = (wave & 3) * 16;   // 4 row-slabs of 16
    const int wn   = (wave >> 2) * 32;  // 2 col-slabs of 32

    v8f acc0 = {}, acc1 = {};

    for (int k0 = 0; k0 < K; k0 += 64) {
        __syncthreads();
        // ---- stage A tile 64x64 fp32 -> bf16 (float4 loads, packed b64 stores)
#pragma unroll
        for (int r = 0; r < 4; ++r) {
            int i = tid + r * 256;                 // quad index, 1024 total
            int m = i >> 4, kq = (i & 15) * 4;
            float4 f = *(const float4*)(A + (long)(tile_m + m) * lda + k0 + kq);
            uint2 u;
            u.x = (unsigned)f2bf(f.x) | ((unsigned)f2bf(f.y) << 16);
            u.y = (unsigned)f2bf(f.z) | ((unsigned)f2bf(f.w) << 16);
            *(uint2*)&shA[m * 64 + kq] = u;
        }
        // ---- stage B tile as [n][k]
        if (BTRANS) {
            // rows contiguous in memory -> async DMA straight into LDS
#pragma unroll
            for (int r = 0; r < 2; ++r) {
                int i = tid + r * 256;             // 16B chunk index, 512 total
                int n = i >> 3, c = i & 7;
                unsigned long long g = (unsigned long long)
                    (Bsrc + (long)(tile_n + n) * ldb + k0 + c * 8);
                async_copy_b128(lds_addr(&shB[n * 64 + c * 8]), g);
            }
            if (k0 + 64 < K)                       // warm next K-tile into L2
                __builtin_prefetch(Bsrc + (long)(tile_n + (tid >> 2)) * ldb + k0 + 64, 0, 1);
            wait_async();
        } else {
            // [K,N] source: transpose-scatter through LDS
#pragma unroll
            for (int r = 0; r < 2; ++r) {
                int i = tid + r * 256;             // 512 groups of 8 elements
                int k = i >> 3, nq = (i & 7) * 8;
                const unsigned short* src = Bsrc + (long)(k0 + k) * ldb + tile_n + nq;
#pragma unroll
                for (int q = 0; q < 8; ++q) shB[(nq + q) * 64 + k] = src[q];
            }
        }
        __syncthreads();

        v16bf a0  = load_fragA(shA + wm * 64,       64);   // K 0..31
        v16bf a1  = load_fragA(shA + wm * 64 + 32,  64);   // K 32..63
        v16bf b00 = load_fragB(shB + (wn +  0) * 64,      64);
        v16bf b01 = load_fragB(shB + (wn + 16) * 64,      64);
        v16bf b10 = load_fragB(shB + (wn +  0) * 64 + 32, 64);
        v16bf b11 = load_fragB(shB + (wn + 16) * 64 + 32, 64);
        acc0 = WMMA_BF16(a0, b00, acc0);
        acc0 = WMMA_BF16(a1, b10, acc0);
        acc1 = WMMA_BF16(a0, b01, acc1);
        acc1 = WMMA_BF16(a1, b11, acc1);
    }

    // ---- epilogue (exact tiling, no guards)
    {
        int n = lane & 15, hi = lane >> 4;
        int row0 = tile_m + wm;
        int c0 = tile_n + wn + n, c1 = c0 + 16;
        float bia0 = bias ? bias[c0] : 0.0f;
        float bia1 = bias ? bias[c1] : 0.0f;
#pragma unroll
        for (int v = 0; v < 8; ++v) {
            int m = row0 + v + hi * 8;
            float x0 = acc0[v] + bia0;
            float x1 = acc1[v] + bia1;
            if (ACT) { x0 = tanhf(x0); x1 = tanhf(x1); }
            D[(long)m * ldd + c0] = x0;
            D[(long)m * ldd + c1] = x1;
        }
    }
}

// ---------------------------------------------------------------------------
// One LSTM time step: gates = xg[:,t,:] + h_prev @ W_hh^T + b_hh; update c,h.
// W_hh pre-converted to bf16 (8 MB -> L2-resident across all 128 steps).
// Grid: H/16 = 64 blocks x 128 threads (wave g = gate g); M = batch = 16.
// K-step 64 -> 2 v_wmma per iteration; W staged by async DMA.
// ---------------------------------------------------------------------------
__global__ __launch_bounds__(128)
void lstm_step(const float* __restrict__ xg, const unsigned short* __restrict__ W_hh,
               const float* __restrict__ b_hh,
               const float* __restrict__ h_prev, float* __restrict__ h_next,
               float* __restrict__ c_state, float* __restrict__ h_seq, int t)
{
    __shared__ __attribute__((aligned(16))) unsigned short shH[16 * 64];
    __shared__ __attribute__((aligned(16))) unsigned short shW[4 * 16 * 64];
    __shared__ float shG[4 * 16 * 16];

    const int j0   = blockIdx.x * 16;
    const int tid  = threadIdx.x;
    const int wave = tid >> 5;       // gate index (i,f,g,o)
    const int lane = tid & 31;

    v8f acc = {};
    for (int k0 = 0; k0 < HH; k0 += 64) {
        __syncthreads();
#pragma unroll
        for (int r = 0; r < 2; ++r) {                     // h tile 16x64 f32->bf16
            int i = tid + r * 128;
            int m = i >> 4, kq = (i & 15) * 4;
            float4 f = *(const float4*)(h_prev + m * HH + k0 + kq);
            uint2 u;
            u.x = (unsigned)f2bf(f.x) | ((unsigned)f2bf(f.y) << 16);
            u.y = (unsigned)f2bf(f.z) | ((unsigned)f2bf(f.w) << 16);
            *(uint2*)&shH[m * 64 + kq] = u;
        }
#pragma unroll
        for (int cc = 0; cc < 4; ++cc) {                  // W rows, async DMA
            int i = lane + cc * 32;                       // 128 chunks of 16B
            int n = i >> 3, ch = i & 7;
            unsigned long long g = (unsigned long long)
                (W_hh + (long)(wave * HH + j0 + n) * HH + k0 + ch * 8);
            async_copy_b128(lds_addr(&shW[wave * 1024 + n * 64 + ch * 8]), g);
        }
        wait_async();
        __syncthreads();
        v16bf a0 = load_fragA(shH, 64);
        v16bf a1 = load_fragA(shH + 32, 64);
        v16bf b0 = load_fragB(shW + wave * 1024, 64);
        v16bf b1 = load_fragB(shW + wave * 1024 + 32, 64);
        acc = WMMA_BF16(a0, b0, acc);
        acc = WMMA_BF16(a1, b1, acc);
    }

    {   // spill gate tile to LDS in [m][n] order
        int n = lane & 15, hi = lane >> 4;
#pragma unroll
        for (int v = 0; v < 8; ++v)
            shG[wave * 256 + (v + hi * 8) * 16 + n] = acc[v];
    }
    __syncthreads();

    // fused LSTM cell: 16x16 (b, j) elements over 128 threads
    for (int i = tid; i < 256; i += 128) {
        int b = i >> 4, jl = i & 15;
        int j = j0 + jl;
        long xb = ((long)b * TT + t) * (4L * HH);
        float gi = shG[0 * 256 + b * 16 + jl] + xg[xb + 0 * HH + j] + b_hh[0 * HH + j];
        float gf = shG[1 * 256 + b * 16 + jl] + xg[xb + 1 * HH + j] + b_hh[1 * HH + j];
        float gg = shG[2 * 256 + b * 16 + jl] + xg[xb + 2 * HH + j] + b_hh[2 * HH + j];
        float go = shG[3 * 256 + b * 16 + jl] + xg[xb + 3 * HH + j] + b_hh[3 * HH + j];
        float c  = c_state[b * HH + j];
        float si = 1.0f / (1.0f + __expf(-gi));
        float sf = 1.0f / (1.0f + __expf(-gf));
        float so = 1.0f / (1.0f + __expf(-go));
        float cn = sf * c + si * tanhf(gg);
        float hn = so * tanhf(cn);
        c_state[b * HH + j] = cn;
        h_next[b * HH + j]  = hn;
        h_seq[((long)b * TT + t) * HH + j] = hn;
    }
}

// ---------------------------------------------------------------------------
// Elementwise helpers
// ---------------------------------------------------------------------------
__global__ void cvt_f32_bf16(const float* __restrict__ src,
                             unsigned short* __restrict__ dst, long npairs)
{
    long i = (long)blockIdx.x * blockDim.x + threadIdx.x;
    if (i < npairs) {
        float2 f = *(const float2*)(src + i * 2);
        *(unsigned*)(dst + i * 2) =
            (unsigned)f2bf(f.x) | ((unsigned)f2bf(f.y) << 16);
    }
}

__global__ void embed_gather(const int* __restrict__ y, const float* __restrict__ emb,
                             float* __restrict__ out)
{
    long i = (long)blockIdx.x * blockDim.x + threadIdx.x;   // over B*T*E
    long bt = i / EE;
    int  e  = (int)(i % EE);
    out[i] = emb[(long)y[bt] * EE + e];
}

__global__ void init_state(const float* __restrict__ h0, const float* __restrict__ c0,
                           float* __restrict__ h, float* __restrict__ c, int n)
{
    int i = blockIdx.x * blockDim.x + threadIdx.x;
    if (i < n) { h[i] = h0[i]; c[i] = c0[i]; }
}

__global__ __launch_bounds__(128)
void softmax_mask(float* __restrict__ scores, const unsigned char* __restrict__ mask)
{
    int bt = blockIdx.x;              // b*T + t
    int b  = bt / TT;
    int s  = threadIdx.x;             // S = 128 threads
    float* row = scores + (long)bt * SS;
    float x = row[s];
    if (!mask[b * SS + s]) x = -1.0e9f;

    __shared__ float red[4];
    float m = x;
    for (int o = 16; o > 0; o >>= 1) m = fmaxf(m, __shfl_xor(m, o, 32));
    if ((s & 31) == 0) red[s >> 5] = m;
    __syncthreads();
    m = fmaxf(fmaxf(red[0], red[1]), fmaxf(red[2], red[3]));
    __syncthreads();

    float e = __expf(x - m);
    float sum = e;
    for (int o = 16; o > 0; o >>= 1) sum += __shfl_xor(sum, o, 32);
    if ((s & 31) == 0) red[s >> 5] = sum;
    __syncthreads();
    sum = red[0] + red[1] + red[2] + red[3];
    row[s] = e / sum;
}

__global__ void concat_oc(const float* __restrict__ o, const float* __restrict__ x,
                          float* __restrict__ cat)
{
    long i = (long)blockIdx.x * blockDim.x + threadIdx.x;   // over BT*2H
    long r = i / (2 * HH);
    int  c = (int)(i % (2 * HH));
    cat[i] = (c < HH) ? o[r * HH + c] : x[r * HH + (c - HH)];
}

// ---------------------------------------------------------------------------
// Host-side orchestration
// ---------------------------------------------------------------------------
extern "C" void kernel_launch(void* const* d_in, const int* in_sizes, int n_in,
                              void* d_out, int out_size, void* d_ws, size_t ws_size,
                              hipStream_t stream)
{
    (void)in_sizes; (void)n_in; (void)out_size; (void)ws_size;

    const int*   y_in  = (const int*)d_in[0];
    const float* h0    = (const float*)d_in[1];
    const float* c0    = (const float*)d_in[2];
    const float* enc   = (const float*)d_in[3];
    const unsigned char* emask = (const unsigned char*)d_in[4];
    const float* embed = (const float*)d_in[5];
    const float* W_ih0 = (const float*)d_in[6];
    const float* W_hh0 = (const float*)d_in[7];
    const float* b_ih0 = (const float*)d_in[8];
    const float* b_hh0 = (const float*)d_in[9];
    const float* W_ih1 = (const float*)d_in[10];
    const float* W_hh1 = (const float*)d_in[11];
    const float* b_ih1 = (const float*)d_in[12];
    const float* b_hh1 = (const float*)d_in[13];
    const float* fc_W  = (const float*)d_in[14];
    const float* fc_b  = (const float*)d_in[15];
    const float* out_W = (const float*)d_in[16];
    const float* out_b = (const float*)d_in[17];
    float* logits = (float*)d_out;

    // ---- fp32 workspace (float units), with reuse
    float* wsf = (float*)d_ws;
    float* emb_buf = wsf;                          // 1,048,576
    float* xg      = wsf + 1048576;                // 8,388,608
    float* out0    = wsf + 1048576 + 8388608;      // 2,097,152
    float* out1    = out0 + 2097152;               // 2,097,152
    float* hA      = out1 + 2097152;               // 16,384
    float* hB      = hA + BB * HH;
    float* cS      = hB + BB * HH;
    float* scores  = emb_buf;                      // reuse (emb dead)
    float* ctx     = xg;                           // reuse (xg dead)
    float* cat     = xg + 2097152;
    float* ht      = out0;                         // reuse (out0 dead)
    const long nfloat = 13680640;                  // total fp32 region

    // ---- bf16 weight workspace (ushort units)
    unsigned short* wsb = (unsigned short*)(wsf + nfloat);
    unsigned short* bWih0 = wsb;                   // 4096*512
    unsigned short* bWhh0 = bWih0 + 4096L * 512;   // 4096*1024
    unsigned short* bWih1 = bWhh0 + 4096L * 1024;
    unsigned short* bWhh1 = bWih1 + 4096L * 1024;
    unsigned short* bFcW  = bWhh1 + 4096L * 1024;  // 1024*2048
    unsigned short* bEnc  = bFcW  + 1024L * 2048;  // 16*128*1024
    unsigned short* bOutW = bEnc  + (long)BB * SS * HH;  // 32000*1024

    const int BT = BB * TT;   // 2048

    // 0) one-pass fp32 -> bf16 conversion of every B operand
    auto cvt = [&](const float* s, unsigned short* d, long n) {
        cvt_f32_bf16<<<(unsigned)((n / 2 + 255) / 256), 256, 0, stream>>>(s, d, n / 2);
    };
    cvt(W_ih0, bWih0, 4096L * 512);
    cvt(W_hh0, bWhh0, 4096L * 1024);
    cvt(W_ih1, bWih1, 4096L * 1024);
    cvt(W_hh1, bWhh1, 4096L * 1024);
    cvt(fc_W,  bFcW,  1024L * 2048);
    cvt(enc,   bEnc,  (long)BB * SS * HH);
    cvt(out_W, bOutW, (long)VV * HH);

    // 1) embedding gather
    embed_gather<<<(BT * EE) / 256, 256, 0, stream>>>(y_in, embed, emb_buf);

    // 2) xg0 = emb @ W_ih0^T + b_ih0       [2048 x 4096], K=512
    gemm_bf16<1, 0><<<dim3(4096 / 64, BT / 64, 1), 256, 0, stream>>>(
        emb_buf, bWih0, b_ih0, xg, BT, 4 * HH, EE, EE, EE, 4 * HH, 0, 0, 0);

    // 3) LSTM layer 0 scan (double-buffered h)
    init_state<<<(BB * HH) / 256, 256, 0, stream>>>(h0, c0, hA, cS, BB * HH);
    for (int t = 0; t < TT; ++t) {
        float* hp = (t & 1) ? hB : hA;
        float* hn = (t & 1) ? hA : hB;
        lstm_step<<<HH / 16, 128, 0, stream>>>(xg, bWhh0, b_hh0, hp, hn, cS, out0, t);
    }

    // 4) xg1 = out0 @ W_ih1^T + b_ih1      [2048 x 4096], K=1024
    gemm_bf16<1, 0><<<dim3(4096 / 64, BT / 64, 1), 256, 0, stream>>>(
        out0, bWih1, b_ih1, xg, BT, 4 * HH, HH, HH, HH, 4 * HH, 0, 0, 0);

    // 5) LSTM layer 1 scan
    init_state<<<(BB * HH) / 256, 256, 0, stream>>>(h0 + BB * HH, c0 + BB * HH,
                                                    hA, cS, BB * HH);
    for (int t = 0; t < TT; ++t) {
        float* hp = (t & 1) ? hB : hA;
        float* hn = (t & 1) ? hA : hB;
        lstm_step<<<HH / 16, 128, 0, stream>>>(xg, bWhh1, b_hh1, hp, hn, cS, out1, t);
    }

    // 6) attention scores = out1 @ enc^T    per batch: [128 x 128], K=1024
    gemm_bf16<1, 0><<<dim3(SS / 64, TT / 64, BB), 256, 0, stream>>>(
        out1, bEnc, nullptr, scores, TT, SS, HH, HH, HH, SS,
        (long)TT * HH, (long)SS * HH, (long)TT * SS);

    // 7) masked softmax over S
    softmax_mask<<<BT, SS, 0, stream>>>(scores, emask);

    // 8) ctx = attn @ enc                   per batch: [128 x 1024], K=128
    gemm_bf16<0, 0><<<dim3(HH / 64, TT / 64, BB), 256, 0, stream>>>(
        scores, bEnc, nullptr, ctx, TT, HH, SS, SS, HH, HH,
        (long)TT * SS, (long)SS * HH, (long)TT * HH);

    // 9) cat = [out1, ctx]                  [2048 x 2048]
    concat_oc<<<(BT * 2 * HH) / 256, 256, 0, stream>>>(out1, ctx, cat);

    // 10) h_t = tanh(cat @ fc_W^T + fc_b)   [2048 x 1024], K=2048
    gemm_bf16<1, 1><<<dim3(HH / 64, BT / 64, 1), 256, 0, stream>>>(
        cat, bFcW, fc_b, ht, BT, HH, 2 * HH, 2 * HH, 2 * HH, HH, 0, 0, 0);

    // 11) logits = h_t @ out_W^T + out_b    [2048 x 32000], K=1024 (dominant;
    //     bf16 out_W = 65 MB -> L2-resident on 192 MB L2, HBM reads halved)
    gemm_bf16<1, 0><<<dim3(VV / 64, BT / 64, 1), 256, 0, stream>>>(
        ht, bOutW, out_b, logits, BT, VV, HH, HH, HH, VV, 0, 0, 0);
}
